// StateSpaceModel_78537771975093
// MI455X (gfx1250) — matrix operand order repaired and checked by reading, not compiled, not run on hardware
//
#include <hip/hip_runtime.h>
#include <hip/hip_bf16.h>

// ---------------------------------------------------------------------------
// SSM scan on MI455X (gfx1250): chunked parallel scan.
//  - All matmuls: v_wmma_f32_16x16x32_f16 (f16 operands, f32 accumulation).
//  - LDS staging of resident 256x256 f16 matrices: Tensor Data Mover
//    (tensor_load_to_lds + s_wait_tensorcnt), no VGPR round-trip.
// ---------------------------------------------------------------------------

#define BSZ    8
#define LEN    8192
#define DIM    256
#define LC     128          // chunk length
#define NCHUNK (LEN / LC)   // 64 chunks

typedef __attribute__((ext_vector_type(16))) _Float16 v16h;
typedef __attribute__((ext_vector_type(8)))  float    v8f;
typedef __attribute__((ext_vector_type(4)))  unsigned int v4u;
typedef __attribute__((ext_vector_type(8)))  int      v8i;
typedef __attribute__((ext_vector_type(4)))  int      v4i;

__device__ __forceinline__ v8f wmma_f16(v16h a, v16h b, v8f c) {
  // (neg_a, A, neg_b, B, c_mod, C, reuse_a, reuse_b)
  return __builtin_amdgcn_wmma_f32_16x16x32_f16(false, a, false, b,
                                                (short)0, c, false, false);
}

// A-operand (16x32 f16) K position for VGPR-pair v (0..7) and lane group g.
__device__ __forceinline__ int afrag_kbase(int v, int g) {
  return ((v < 4) ? (2 * v) : (16 + 2 * (v - 4))) + g * 8;
}

// ---------------------------------------------------------------------------
// Tensor Data Mover: DMA a rows x cols f16 tile (row-major, stride = cols)
// from global memory into LDS at byte offset lds_byte_off.
// D# built per CDNA5 ISA 8.3/8.4: group0 = {count, lds_addr, global_addr,
// type=2}; group1 = {data_size=2B, tensor dims, tile dims, dim0 stride}.
// ---------------------------------------------------------------------------
__device__ __forceinline__ void tdm_load_f16_2d(const _Float16* gsrc,
                                                unsigned lds_byte_off,
                                                unsigned rows, unsigned cols) {
  unsigned long long ga = (unsigned long long)gsrc;
  v4u g0;
  g0[0] = 1u;                                   // count=1 (valid), user mode
  g0[1] = lds_byte_off;                         // lds_addr (bytes)
  g0[2] = (unsigned)ga;                         // global_addr[31:0]
  g0[3] = (unsigned)(ga >> 32) | (2u << 30);    // global_addr[56:32] | type=2
  v8i g1;
  g1[0] = (1 << 16);                            // workgroup_mask=0, data_size=1 (2B)
  g1[1] = (int)((cols & 0xFFFFu) << 16);        // tensor_dim0[15:0]
  g1[2] = (int)(((cols >> 16) & 0xFFFFu) | ((rows & 0xFFFFu) << 16)); // d0 hi | d1 lo
  g1[3] = (int)(((rows >> 16) & 0xFFFFu) | ((cols & 0xFFFFu) << 16)); // d1 hi | tile_dim0
  g1[4] = (int)(rows & 0xFFFFu);                // tile_dim1 (tile_dim2=0)
  g1[5] = (int)cols;                            // tensor_dim0_stride[31:0]
  g1[6] = 0;
  g1[7] = 0;
  v4i z4 = {0, 0, 0, 0};
#if defined(__clang_major__) && (__clang_major__ >= 23)
  v8i z8 = {0, 0, 0, 0, 0, 0, 0, 0};
  __builtin_amdgcn_tensor_load_to_lds(g0, g1, z4, z4, z8, 0);
#else
  __builtin_amdgcn_tensor_load_to_lds(g0, g1, z4, z4, 0);
#endif
}

// Issue TDM from wave 0 only, wait for completion, then block barrier.
__device__ __forceinline__ void stage_lds_tdm(const _Float16* gsrc,
                                              unsigned lds_byte_off,
                                              unsigned rows, unsigned cols) {
  if (threadIdx.x < 32u) {
    tdm_load_f16_2d(gsrc, lds_byte_off, rows, cols);
    __builtin_amdgcn_s_wait_tensorcnt((short)0);
  }
  __syncthreads();
}

// ---------------------------------------------------------------------------
// K0: precision prep: Ah = f16(A); Cth[k][n] = f16(C[n][k]).
// ---------------------------------------------------------------------------
__global__ void k_prep(const float* __restrict__ A, const float* __restrict__ C,
                       _Float16* __restrict__ Ah, _Float16* __restrict__ Cth) {
  const int i = blockIdx.x * blockDim.x + threadIdx.x;   // 0 .. DIM*DIM-1
  Ah[i] = (_Float16)A[i];
  const int k = i >> 8, n = i & 255;
  Cth[k * DIM + n] = (_Float16)C[n * DIM + k];
}

// ---------------------------------------------------------------------------
// K1: U = X @ B^T  (M=BSZ*LEN=65536, K=DIM, N=DIM), U stored f16.
// block = 256 threads (8 waves), each wave one 16x16 tile; grid=(512,16).
// ---------------------------------------------------------------------------
__global__ void k_u_gemm(const float* __restrict__ x, const float* __restrict__ B,
                         _Float16* __restrict__ U) {
  extern __shared__ _Float16 smem[];                 // Bt panel [256][16]
  const int n0 = blockIdx.y * 16;
  for (int i = threadIdx.x; i < DIM * 16; i += blockDim.x) {
    int k = i >> 4, nn = i & 15;
    smem[k * 16 + nn] = (_Float16)B[(n0 + nn) * DIM + k];   // B^T[k][n] = B[n][k]
  }
  __syncthreads();

  const int wave = threadIdx.x >> 5;
  const int lane = threadIdx.x & 31;
  const int g    = lane >> 4;
  const int m0   = (blockIdx.x * 8 + wave) * 16;
  const int mrow = m0 + (lane & 15);

  v8f acc = {};
#pragma unroll
  for (int kk = 0; kk < DIM; kk += 32) {
    v16h a;
#pragma unroll
    for (int v = 0; v < 8; ++v) {
      int k = kk + afrag_kbase(v, g);                 // always even
      const float2 xv = *(const float2*)(x + (size_t)mrow * DIM + k);
      a[2 * v]     = (_Float16)xv.x;
      a[2 * v + 1] = (_Float16)xv.y;
    }
    v16h b;
    for (int s = 0; s < 16; ++s) b[s] = smem[(kk + afrag_kbase(s >> 1, g) + (s & 1)) * 16 + (lane & 15)];
    acc = wmma_f16(a, b, acc);
  }
#pragma unroll
  for (int r = 0; r < 8; ++r) {
    int m = m0 + ((lane < 16) ? r : r + 8);
    U[(size_t)m * DIM + n0 + (lane & 15)] = (_Float16)acc[r];
  }
}

// ---------------------------------------------------------------------------
// K2: matrix powers  P[k-1] = A^k (f16), k = 1..LC.  Single block, 16 waves,
// A resident in LDS (TDM-staged); 127 dependent 256x256x256 GEMMs.
// ---------------------------------------------------------------------------
__global__ void k_powers(const _Float16* __restrict__ Ah, _Float16* __restrict__ P) {
  extern __shared__ _Float16 Alds[];                 // [256][256] = 128 KB
  stage_lds_tdm(Ah, 0, DIM, DIM);
  for (int i = threadIdx.x; i < DIM * DIM; i += blockDim.x)
    P[i] = Ah[i];                                    // P[0] = A^1
  __threadfence();
  __syncthreads();

  const int lane = threadIdx.x & 31, g = lane >> 4;
  const int n0 = (threadIdx.x >> 5) * 16;            // wave -> N tile

  for (int step = 1; step < LC; ++step) {
    const _Float16* Pprev = P + (size_t)(step - 1) * DIM * DIM;
    _Float16*       Pnew  = P + (size_t)step * DIM * DIM;
#pragma unroll 1
    for (int mt = 0; mt < 16; ++mt) {
      const int m0 = mt * 16, m = m0 + (lane & 15);
      v8f acc = {};
#pragma unroll
      for (int kk = 0; kk < DIM; kk += 32) {
        v16h a;
#pragma unroll
        for (int v = 0; v < 8; ++v) {
          int k = kk + afrag_kbase(v, g);
          a[2 * v]     = Pprev[m * DIM + k];
          a[2 * v + 1] = Pprev[m * DIM + k + 1];
        }
        v16h b;
        for (int s = 0; s < 16; ++s) b[s] = Alds[(kk + afrag_kbase(s >> 1, g) + (s & 1)) * DIM + n0 + (lane & 15)];
        acc = wmma_f16(a, b, acc);
      }
#pragma unroll
      for (int r = 0; r < 8; ++r) {
        int mm = m0 + ((lane < 16) ? r : r + 8);
        Pnew[mm * DIM + n0 + (lane & 15)] = (_Float16)acc[r];
      }
    }
    __threadfence();
    __syncthreads();
  }
}

// ---------------------------------------------------------------------------
// K3: per-chunk local scan (zero init): s_t = s_{t-1} @ A + u_t.
// grid = NCHUNK blocks, 512 threads = 16 waves (one N tile each).
// A TDM-staged into LDS; state (16x256 f16, rows 8..15 zero pad) ping-pongs
// through LDS. Local states -> d_out (scratch); chunk finals -> Fc.
// ---------------------------------------------------------------------------
__global__ void k_local_scan(const _Float16* __restrict__ Ah, const _Float16* __restrict__ U,
                             float* __restrict__ Sloc, float* __restrict__ Fc) {
  extern __shared__ _Float16 smem[];
  _Float16* Alds = smem;                             // 256*256 (TDM target, offset 0)
  _Float16* st   = smem + DIM * DIM;                 // 2 * 16*256 (double buffer)
  for (int i = threadIdx.x; i < 2 * 16 * DIM; i += blockDim.x)
    st[i] = (_Float16)0.0f;
  stage_lds_tdm(Ah, 0, DIM, DIM);                    // includes __syncthreads()

  const int c = blockIdx.x;
  const int lane = threadIdx.x & 31, g = lane >> 4;
  const int n0 = (threadIdx.x >> 5) * 16, nn = lane & 15;
  int rd = 0;

  for (int tl = 0; tl < LC; ++tl) {
    const int t = c * LC + tl;
    if (tl + 8 < LC)                                 // stream U ahead of use
      __builtin_prefetch((const void*)(U + ((size_t)(lane & 7) * LEN + t + 8) * DIM), 0, 0);

    v8f acc;                                         // C init = u_t (pad rows 0)
#pragma unroll
    for (int r = 0; r < 8; ++r) {
      int m = (lane < 16) ? r : r + 8;
      acc[r] = (m < BSZ) ? (float)U[((size_t)m * LEN + t) * DIM + n0 + nn] : 0.0f;
    }
    const _Float16* srd = st + rd * 16 * DIM;
#pragma unroll
    for (int kk = 0; kk < DIM; kk += 32) {
      v16h a;
#pragma unroll
      for (int v = 0; v < 8; ++v) {
        int k = kk + afrag_kbase(v, g);
        a[2 * v]     = srd[(lane & 15) * DIM + k];
        a[2 * v + 1] = srd[(lane & 15) * DIM + k + 1];
      }
      v16h b;
      for (int s = 0; s < 16; ++s) b[s] = Alds[(kk + afrag_kbase(s >> 1, g) + (s & 1)) * DIM + n0 + nn];
      acc = wmma_f16(a, b, acc);
    }
    _Float16* swr = st + (rd ^ 1) * 16 * DIM;
#pragma unroll
    for (int r = 0; r < 8; ++r) {
      int m = (lane < 16) ? r : r + 8;
      swr[m * DIM + n0 + nn] = (_Float16)acc[r];
      if (m < BSZ) {
        Sloc[((size_t)m * LEN + t) * DIM + n0 + nn] = acc[r];
        if (tl == LC - 1)
          Fc[((size_t)c * BSZ + m) * DIM + n0 + nn] = acc[r];
      }
    }
    rd ^= 1;
    __syncthreads();
  }
}

// ---------------------------------------------------------------------------
// K4: carry scan across chunk boundaries: carry_{c+1} = carry_c @ A^LC + F_c.
// 63 dependent small GEMMs; single block; A^LC TDM-staged into LDS.
// ---------------------------------------------------------------------------
__global__ void k_carry_scan(const _Float16* __restrict__ Palc,  // = P[LC-1] = A^LC
                             const float* __restrict__ Fc, float* __restrict__ carries) {
  extern __shared__ _Float16 smem[];
  _Float16* Alds = smem;
  _Float16* st   = smem + DIM * DIM;
  for (int i = threadIdx.x; i < 2 * 16 * DIM; i += blockDim.x) st[i] = (_Float16)0.0f;
  for (int i = threadIdx.x; i < BSZ * DIM; i += blockDim.x) carries[i] = 0.0f;
  stage_lds_tdm(Palc, 0, DIM, DIM);                  // includes __syncthreads()

  const int lane = threadIdx.x & 31, g = lane >> 4;
  const int n0 = (threadIdx.x >> 5) * 16, nn = lane & 15;
  int rd = 0;

  for (int c = 0; c < NCHUNK - 1; ++c) {
    v8f acc;                                         // C init = F_c
#pragma unroll
    for (int r = 0; r < 8; ++r) {
      int m = (lane < 16) ? r : r + 8;
      acc[r] = (m < BSZ) ? Fc[((size_t)c * BSZ + m) * DIM + n0 + nn] : 0.0f;
    }
    const _Float16* srd = st + rd * 16 * DIM;
#pragma unroll
    for (int kk = 0; kk < DIM; kk += 32) {
      v16h a;
#pragma unroll
      for (int v = 0; v < 8; ++v) {
        int k = kk + afrag_kbase(v, g);
        a[2 * v]     = srd[(lane & 15) * DIM + k];
        a[2 * v + 1] = srd[(lane & 15) * DIM + k + 1];
      }
      v16h b;
      for (int s = 0; s < 16; ++s) b[s] = Alds[(kk + afrag_kbase(s >> 1, g) + (s & 1)) * DIM + n0 + nn];
      acc = wmma_f16(a, b, acc);
    }
    _Float16* swr = st + (rd ^ 1) * 16 * DIM;
#pragma unroll
    for (int r = 0; r < 8; ++r) {
      int m = (lane < 16) ? r : r + 8;
      swr[m * DIM + n0 + nn] = (_Float16)acc[r];
      if (m < BSZ)
        carries[((size_t)(c + 1) * BSZ + m) * DIM + n0 + nn] = acc[r];
    }
    rd ^= 1;
    __syncthreads();
  }
}

// ---------------------------------------------------------------------------
// K5: fix-up + output.  For each t (one block each):
//   state_t = S_local_t + carry_{t/LC} @ A^{(t%LC)+1}
//   out_t   = state_t @ C^T          (C^T f16 TDM-staged into LDS)
// ---------------------------------------------------------------------------
__global__ void k_output(const _Float16* __restrict__ Cth, const _Float16* __restrict__ P,
                         const float* __restrict__ carries, float* __restrict__ out) {
  extern __shared__ _Float16 smem[];
  _Float16* Ct = smem;                               // C^T[k][n], TDM target
  _Float16* st = smem + DIM * DIM;                   // 16x256 state
  stage_lds_tdm(Cth, 0, DIM, DIM);

  const int t = blockIdx.x;
  const int tl = t % LC;
  const _Float16* Pk    = P + (size_t)tl * DIM * DIM;        // A^{tl+1}
  const float*    carry = carries + (size_t)(t / LC) * BSZ * DIM;
  const int lane = threadIdx.x & 31, g = lane >> 4;
  const int n0 = (threadIdx.x >> 5) * 16, nn = lane & 15;

  // phase 1: state = S_local (in `out` scratch) + carry @ A^{tl+1}
  v8f acc;
#pragma unroll
  for (int r = 0; r < 8; ++r) {
    int m = (lane < 16) ? r : r + 8;
    acc[r] = (m < BSZ) ? out[((size_t)m * LEN + t) * DIM + n0 + nn] : 0.0f;
  }
#pragma unroll
  for (int kk = 0; kk < DIM; kk += 32) {
    v16h a;
    const int m = lane & 15;
#pragma unroll
    for (int v = 0; v < 8; ++v) {
      int k = kk + afrag_kbase(v, g);
      a[2 * v]     = (_Float16)((m < BSZ) ? carry[m * DIM + k]     : 0.0f);
      a[2 * v + 1] = (_Float16)((m < BSZ) ? carry[m * DIM + k + 1] : 0.0f);
    }
    v16h b;
    for (int s = 0; s < 16; ++s) b[s] = Pk[(kk + afrag_kbase(s >> 1, g) + (s & 1)) * DIM + n0 + nn];
    acc = wmma_f16(a, b, acc);
  }
#pragma unroll
  for (int r = 0; r < 8; ++r) {
    int m = (lane < 16) ? r : r + 8;
    st[m * DIM + n0 + nn] = (m < BSZ) ? (_Float16)acc[r] : (_Float16)0.0f;
  }
  __syncthreads();

  // phase 2: out = state @ C^T
  v8f o = {};
#pragma unroll
  for (int kk = 0; kk < DIM; kk += 32) {
    v16h a;
#pragma unroll
    for (int v = 0; v < 8; ++v) {
      int k = kk + afrag_kbase(v, g);
      a[2 * v]     = st[(lane & 15) * DIM + k];
      a[2 * v + 1] = st[(lane & 15) * DIM + k + 1];
    }
    v16h b;
    for (int s = 0; s < 16; ++s) b[s] = Ct[(kk + afrag_kbase(s >> 1, g) + (s & 1)) * DIM + n0 + nn];
    o = wmma_f16(a, b, o);
  }
#pragma unroll
  for (int r = 0; r < 8; ++r) {
    int m = (lane < 16) ? r : r + 8;
    if (m < BSZ) out[((size_t)m * LEN + t) * DIM + n0 + nn] = o[r];
  }
}

// ---------------------------------------------------------------------------
// Launch
// ---------------------------------------------------------------------------
static inline void set_lds(const void* f, int bytes) {
  (void)hipFuncSetAttribute(f, hipFuncAttributeMaxDynamicSharedMemorySize, bytes);
}

extern "C" void kernel_launch(void* const* d_in, const int* in_sizes, int n_in,
                              void* d_out, int out_size, void* d_ws, size_t ws_size,
                              hipStream_t stream) {
  const float* x = (const float*)d_in[0];
  const float* A = (const float*)d_in[1];
  const float* B = (const float*)d_in[2];
  const float* C = (const float*)d_in[3];
  float* out = (float*)d_out;

  char* ws = (char*)d_ws;
  _Float16* U   = (_Float16*)ws;                                       // 32 MB: x @ B^T (f16)
  _Float16* P   = (_Float16*)(ws + (size_t)32 * 1024 * 1024);          // 16 MB: A^1..A^128 (f16)
  float*    Fc  = (float*)   (ws + (size_t)48 * 1024 * 1024);          // 512 KB: chunk finals
  float*    Cr  = (float*)   (ws + (size_t)48 * 1024 * 1024 + 512 * 1024); // 512 KB: carries
  _Float16* Ah  = (_Float16*)(ws + (size_t)49 * 1024 * 1024);          // 128 KB: f16(A)
  _Float16* Cth = (_Float16*)(ws + (size_t)49 * 1024 * 1024 + 128 * 1024); // 128 KB: f16(C^T)

  const int lds_bpanel = DIM * 16 * (int)sizeof(_Float16);                 //   8 KB
  const int lds_pow    = DIM * DIM * (int)sizeof(_Float16);                // 128 KB
  const int lds_scan   = (DIM * DIM + 2 * 16 * DIM) * (int)sizeof(_Float16); // 144 KB
  const int lds_out    = (DIM * DIM + 16 * DIM) * (int)sizeof(_Float16);   // 136 KB

  set_lds(reinterpret_cast<const void*>(&k_powers),     lds_pow);
  set_lds(reinterpret_cast<const void*>(&k_local_scan), lds_scan);
  set_lds(reinterpret_cast<const void*>(&k_carry_scan), lds_scan);
  set_lds(reinterpret_cast<const void*>(&k_output),     lds_out);

  // K0: f16 copies of A and C^T (TDM sources)
  k_prep<<<DIM, DIM, 0, stream>>>(A, C, Ah, Cth);
  // K1: U = X @ B^T (parallel)            K2: A powers (depth 128)
  dim3 g1(512, 16);
  k_u_gemm<<<g1, 256, lds_bpanel, stream>>>(x, B, U);
  k_powers<<<1, 512, lds_pow, stream>>>(Ah, P);
  // K3: local scans (64 parallel chains of depth 128)
  k_local_scan<<<NCHUNK, 512, lds_scan, stream>>>(Ah, U, out, Fc);
  // K4: carry scan (depth 64)
  k_carry_scan<<<1, 512, lds_scan, stream>>>(P + (size_t)(LC - 1) * DIM * DIM, Fc, Cr);
  // K5: fix-up + C^T projection (fully parallel)
  k_output<<<LEN, 512, lds_out, stream>>>(Cth, P, Cr, out);
}
